// NasCiteseer_41480794145001
// MI455X (gfx1250) — matrix operand order repaired and checked by reading, not compiled, not run on hardware
//
#include <hip/hip_runtime.h>
#include <hip/hip_bf16.h>

typedef _Float16 v16h __attribute__((ext_vector_type(16)));
typedef _Float16 v8h  __attribute__((ext_vector_type(8)));
typedef float    v8f  __attribute__((ext_vector_type(8)));

#define GN    50000
#define F_IN  500
#define HID   64
#define HEADS 6
#define CH    64
#define HC    384   /* HEADS*CH */
#define NC    32

/* ---------------- WMMA GEMM ----------------
 * C[M,Nc] = A[M,Kp](f16, row-major, stride lda) x Bt[Nc,Kp](f16, row-major = B^T)
 *           (+ bias[Nc]) (+ old C if accum)
 * One wave (32 threads) per 16x16 output tile. grid = (Nc/16, M/16).
 * EXEC is all-ones everywhere (no divergence) as WMMA requires.
 */
__global__ __launch_bounds__(32)
void wmma_gemm_k(const _Float16* __restrict__ A,
                 const _Float16* __restrict__ Bt,
                 const float* __restrict__ bias,
                 float* __restrict__ C,
                 int Kp, int lda, int ldb, int ldc, int accum)
{
    const int lane = threadIdx.x;        // 0..31
    const int kh   = lane >> 4;          // which K-half this lane holds
    const int idx  = lane & 15;          // row (A) / col (B) within tile
    const size_t rowA = (size_t)(blockIdx.y * 16 + idx) * (size_t)lda;
    const size_t rowB = (size_t)(blockIdx.x * 16 + idx) * (size_t)ldb;

    v8f acc = {};
    for (int k0 = 0; k0 < Kp; k0 += 32) {
        // A fragment: lanes 0-15 hold K = k0+{0..7,16..23}, lanes 16-31 K = k0+{8..15,24..31}
        v8h alo = *(const v8h*)(A + rowA + k0 + kh * 8);
        v8h ahi = *(const v8h*)(A + rowA + k0 + 16 + kh * 8);
        // B fragment: lanes 0-15 hold K = k0+{0..7,16..23}, lanes 16-31 K = k0+{8..15,24..31} (col idx)
        v8h blo = *(const v8h*)(Bt + rowB + k0 + kh * 8);
        v8h bhi = *(const v8h*)(Bt + rowB + k0 + 16 + kh * 8);
        v16h a = __builtin_shufflevector(alo, ahi, 0,1,2,3,4,5,6,7,8,9,10,11,12,13,14,15);
        v16h b = __builtin_shufflevector(blo, bhi, 0,1,2,3,4,5,6,7,8,9,10,11,12,13,14,15);
        acc = __builtin_amdgcn_wmma_f32_16x16x32_f16(false, a, false, b,
                                                     (short)0, acc, false, false);
    }

    const int nOut = blockIdx.x * 16 + idx;
    const float bv = bias ? bias[nOut] : 0.0f;
#pragma unroll
    for (int r = 0; r < 8; ++r) {
        // D layout: VGPR r -> M = r (lanes 0-15) / 8+r (lanes 16-31)
        const int mOut = blockIdx.y * 16 + kh * 8 + r;
        const size_t off = (size_t)mOut * (size_t)ldc + nOut;
        float v = acc[r] + bv;
        if (accum) v += C[off];
        C[off] = v;
    }
}

/* ---------- f32 -> f16 with K padding (zero fill) ---------- */
__global__ void conv_f16_k(const float* __restrict__ x, _Float16* __restrict__ y,
                           int K, int Kp, long long total)
{
    long long t = (long long)blockIdx.x * blockDim.x + threadIdx.x;
    if (t >= total) return;
    long long m = t / Kp;
    int k = (int)(t % Kp);
    y[t] = (k < K) ? (_Float16)x[m * K + k] : (_Float16)0.0f;
}

/* ---------- weight transpose+convert: Bt[n,k] = W[rowOff+k, n] ---------- */
__global__ void transpose_w_k(const float* __restrict__ W, _Float16* __restrict__ Bt,
                              int K, int Nc, int rowOff, int Kp, int total)
{
    int t = blockIdx.x * blockDim.x + threadIdx.x;
    if (t >= total) return;   // total = Nc*Kp
    int n = t / Kp, k = t % Kp;
    Bt[t] = (k < K) ? (_Float16)W[(size_t)(rowOff + k) * Nc + n] : (_Float16)0.0f;
}

/* ---------- per-(node,head) attention scores ---------- */
__global__ void att_scores_k(const float* __restrict__ xh,
                             const float* __restrict__ a_src,
                             const float* __restrict__ a_dst,
                             float* __restrict__ ssrc, float* __restrict__ sdst,
                             int total /* N*HEADS */)
{
    int t = blockIdx.x * blockDim.x + threadIdx.x;
    if (t >= total) return;
    int n = t / HEADS, h = t % HEADS;
    const float* xp = xh + (size_t)n * HC + h * CH;
    const float* as = a_src + h * CH;
    const float* ad = a_dst + h * CH;
    float s0 = 0.f, s1 = 0.f;
#pragma unroll 8
    for (int c = 0; c < CH; ++c) { s0 += xp[c] * as[c]; s1 += xp[c] * ad[c]; }
    ssrc[t] = s0; sdst[t] = s1;
}

/* ---------- order-preserving float<->uint map for atomic max ---------- */
__device__ __forceinline__ unsigned fmap(float f) {
    unsigned u = __float_as_uint(f);
    return (u & 0x80000000u) ? ~u : (u | 0x80000000u);
}
__device__ __forceinline__ float funmap(unsigned u) {
    unsigned b = (u & 0x80000000u) ? (u & 0x7FFFFFFFu) : ~u;
    return __uint_as_float(b);
}

__device__ __forceinline__ void get_sd(const int* __restrict__ src,
                                       const int* __restrict__ dst,
                                       int e, int E, int& s, int& d)
{
    if (e < E) { s = src[e]; d = dst[e]; }
    else       { s = e - E;  d = e - E;  }   // self loop
}

/* ---------- pass 1: segment max of leaky_relu(s_src[s]+s_dst[d], 0.2) ---------- */
__global__ void edge_max_k(const int* __restrict__ src, const int* __restrict__ dst,
                           const float* __restrict__ ssrc, const float* __restrict__ sdst,
                           unsigned* __restrict__ mu, int E, int Ntot)
{
    int e = blockIdx.x * blockDim.x + threadIdx.x;
    if (e >= Ntot) return;
    int s, d; get_sd(src, dst, e, E, s, d);
#pragma unroll
    for (int h = 0; h < HEADS; ++h) {
        float v = ssrc[s * HEADS + h] + sdst[d * HEADS + h];
        v = v > 0.f ? v : 0.2f * v;
        atomicMax(&mu[d * HEADS + h], fmap(v));
    }
}

__global__ void finalize_max_k(unsigned* __restrict__ mu, float* __restrict__ m, int total)
{
    int t = blockIdx.x * blockDim.x + threadIdx.x;
    if (t >= total) return;
    m[t] = funmap(mu[t]);    // every segment has a self loop -> always finite
}

/* ---------- pass 2: ex = exp(e - m[dst]); z[dst] += ex ---------- */
__global__ void edge_exp_k(const int* __restrict__ src, const int* __restrict__ dst,
                           const float* __restrict__ ssrc, const float* __restrict__ sdst,
                           const float* __restrict__ m, float* __restrict__ exb,
                           float* __restrict__ z, int E, int Ntot)
{
    int e = blockIdx.x * blockDim.x + threadIdx.x;
    if (e >= Ntot) return;
    int s, d; get_sd(src, dst, e, E, s, d);
#pragma unroll
    for (int h = 0; h < HEADS; ++h) {
        float v = ssrc[s * HEADS + h] + sdst[d * HEADS + h];
        v = v > 0.f ? v : 0.2f * v;
        float ex = __expf(v - m[d * HEADS + h]);
        exb[(size_t)e * HEADS + h] = ex;
        atomicAdd(&z[d * HEADS + h], ex);
    }
}

/* ---------- pass 3: out[dst] += (ex/z[dst]) * xh[src], 4 channels/thread ---------- */
__global__ void edge_scatter_k(const int* __restrict__ src, const int* __restrict__ dst,
                               const float* __restrict__ exb, const float* __restrict__ z,
                               const float* __restrict__ xh, float* __restrict__ go,
                               int E, long long totalT /* Ntot*96 */)
{
    long long t = (long long)blockIdx.x * blockDim.x + threadIdx.x;
    if (t >= totalT) return;
    int e = (int)(t / 96);
    int q = (int)(t % 96);
    int h  = q >> 4;
    int c4 = (q & 15) << 2;
    int s, d; get_sd(src, dst, e, E, s, d);
    float alpha = exb[(size_t)e * HEADS + h] / (z[d * HEADS + h] + 1e-16f);
    const float4 xv = *(const float4*)(xh + (size_t)s * HC + h * CH + c4);
    float* gp = go + (size_t)d * HC + h * CH + c4;
    atomicAdd(gp + 0, alpha * xv.x);
    atomicAdd(gp + 1, alpha * xv.y);
    atomicAdd(gp + 2, alpha * xv.z);
    atomicAdd(gp + 3, alpha * xv.w);
}

/* ---------- h1 = leaky_relu(gat_out + b_gat, 0.01) -> f16 ---------- */
__global__ void bias_lrelu_f16_k(const float* __restrict__ go, const float* __restrict__ bias,
                                 _Float16* __restrict__ y, long long total)
{
    long long t = (long long)blockIdx.x * blockDim.x + threadIdx.x;
    if (t >= total) return;
    int c = (int)(t % HC);
    float v = go[t] + bias[c];
    v = v > 0.f ? v : 0.01f * v;
    y[t] = (_Float16)v;
}

/* ---------- in-place log_softmax over rows of 32 ---------- */
__global__ void log_softmax_k(float* __restrict__ out, int N)
{
    int n = blockIdx.x * blockDim.x + threadIdx.x;
    if (n >= N) return;
    float* row = out + (size_t)n * NC;
    float mx = row[0];
#pragma unroll
    for (int i = 1; i < NC; ++i) mx = fmaxf(mx, row[i]);
    float s = 0.f;
#pragma unroll
    for (int i = 0; i < NC; ++i) s += __expf(row[i] - mx);
    float l = __logf(s) + mx;
#pragma unroll
    for (int i = 0; i < NC; ++i) row[i] = row[i] - l;
}

/* =================== host orchestration =================== */

#define CDIV(a, b) (((a) + (b) - 1) / (b))

/* scratch layout (bytes); aliasing is deliberate & ordered by the stream */
static const size_t o_h     = 0;            /* h f32     N*64*4  = 12.8 MB */
static const size_t o_h16   = 12800000;     /* h f16              6.4 MB  */
static const size_t o_xh    = 19200000;     /* xh f32            76.8 MB  (h1_f16 aliases: 38.4 MB) */
static const size_t o_ssrc  = 96000000;     /* N*6*4 = 1.2 MB */
static const size_t o_sdst  = 97200000;
static const size_t o_m     = 98400000;     /* uint during max, float after */
static const size_t o_z     = 99600000;
static const size_t o_ex    = 100800000;    /* (E+N)*6*4 = 20.4 MB */
static const size_t o_go    = 121200000;    /* gat out f32 76.8 MB (layer-0 in_f16 aliases: 51.2 MB) */
static const size_t o_out   = 198000000;    /* cell out f32 12.8 MB */
static const size_t o_out16 = 210800000;    /* 6.4 MB (layer-1 in_f16 / cls in_f16) */
static const size_t o_w     = 217200000;    /* converted weights, ~180 KB */

static void run_layer(const float* in, int F, int Kp,
                      const float* w_pre, const float* b_pre,
                      const float* w_gat, const float* a_src, const float* a_dst,
                      const float* b_gat, const float* w_lin, const float* b_lin,
                      const int* srcI, const int* dstI, int E,
                      char* ws, _Float16* in16, float* outF, hipStream_t stream)
{
    float*    h    = (float*)(ws + o_h);
    _Float16* h16  = (_Float16*)(ws + o_h16);
    float*    xh   = (float*)(ws + o_xh);
    _Float16* h116 = (_Float16*)(ws + o_xh);   /* aliases xh (xh dead after scatter) */
    float*    ssrc = (float*)(ws + o_ssrc);
    float*    sdst = (float*)(ws + o_sdst);
    unsigned* mu   = (unsigned*)(ws + o_m);
    float*    m    = (float*)(ws + o_m);
    float*    z    = (float*)(ws + o_z);
    float*    exb  = (float*)(ws + o_ex);
    float*    go   = (float*)(ws + o_go);
    _Float16* wpre_t = (_Float16*)(ws + o_w);
    _Float16* wgat_t = (_Float16*)(ws + o_w + 65536);
    _Float16* wlt    = (_Float16*)(ws + o_w + 114688);
    _Float16* wlb    = (_Float16*)(ws + o_w + 122880);

    const int Ntot = E + GN;

    /* h = in @ w_pre + b_pre */
    {
        long long tot = (long long)GN * Kp;
        conv_f16_k<<<CDIV(tot, 256), 256, 0, stream>>>(in, in16, F, Kp, tot);
        transpose_w_k<<<CDIV(HID * Kp, 256), 256, 0, stream>>>(w_pre, wpre_t, F, HID, 0, Kp, HID * Kp);
        wmma_gemm_k<<<dim3(HID / 16, GN / 16), 32, 0, stream>>>(in16, wpre_t, b_pre, h,
                                                                Kp, Kp, Kp, HID, 0);
    }
    /* xh = h @ w_gat (no bias) */
    {
        long long tot = (long long)GN * HID;
        conv_f16_k<<<CDIV(tot, 256), 256, 0, stream>>>(h, h16, HID, HID, tot);
        transpose_w_k<<<CDIV(HC * HID, 256), 256, 0, stream>>>(w_gat, wgat_t, HID, HC, 0, HID, HC * HID);
        wmma_gemm_k<<<dim3(HC / 16, GN / 16), 32, 0, stream>>>(h16, wgat_t, nullptr, xh,
                                                               HID, HID, HID, HC, 0);
    }
    /* attention scores, segment softmax, scatter */
    att_scores_k<<<CDIV(GN * HEADS, 256), 256, 0, stream>>>(xh, a_src, a_dst, ssrc, sdst, GN * HEADS);
    hipMemsetAsync(mu, 0, (size_t)GN * HEADS * 4, stream);
    hipMemsetAsync(z, 0, (size_t)GN * HEADS * 4, stream);
    hipMemsetAsync(go, 0, (size_t)GN * HC * 4, stream);
    edge_max_k<<<CDIV(Ntot, 256), 256, 0, stream>>>(srcI, dstI, ssrc, sdst, mu, E, Ntot);
    finalize_max_k<<<CDIV(GN * HEADS, 256), 256, 0, stream>>>(mu, m, GN * HEADS);
    edge_exp_k<<<CDIV(Ntot, 256), 256, 0, stream>>>(srcI, dstI, ssrc, sdst, m, exb, z, E, Ntot);
    {
        long long tt = (long long)Ntot * 96;
        edge_scatter_k<<<CDIV(tt, 256), 256, 0, stream>>>(srcI, dstI, exb, z, xh, go, E, tt);
    }
    /* h1 = lrelu(go + b_gat, 0.01) -> f16  (overwrites xh region) */
    {
        long long tot = (long long)GN * HC;
        bias_lrelu_f16_k<<<CDIV(tot, 256), 256, 0, stream>>>(go, b_gat, h116, tot);
    }
    /* out = h @ w_lin[0:64] + b_lin  ;  out += h1 @ w_lin[64:448] */
    transpose_w_k<<<CDIV(HID * HID, 256), 256, 0, stream>>>(w_lin, wlt, HID, HID, 0, HID, HID * HID);
    wmma_gemm_k<<<dim3(HID / 16, GN / 16), 32, 0, stream>>>(h16, wlt, b_lin, outF,
                                                            HID, HID, HID, HID, 0);
    transpose_w_k<<<CDIV(HID * HC, 256), 256, 0, stream>>>(w_lin, wlb, HC, HID, HID, HC, HID * HC);
    wmma_gemm_k<<<dim3(HID / 16, GN / 16), 32, 0, stream>>>(h116, wlb, nullptr, outF,
                                                            HC, HC, HC, HID, 1);
}

extern "C" void kernel_launch(void* const* d_in, const int* in_sizes, int n_in,
                              void* d_out, int out_size, void* d_ws, size_t ws_size,
                              hipStream_t stream)
{
    const float* x    = (const float*)d_in[0];
    const int*   ei   = (const int*)d_in[1];
    const int    E    = in_sizes[1] / 2;
    const int*   srcI = ei;
    const int*   dstI = ei + E;

    const float* w_pre0 = (const float*)d_in[3];
    const float* b_pre0 = (const float*)d_in[4];
    const float* w_gat0 = (const float*)d_in[5];
    const float* a_src0 = (const float*)d_in[6];
    const float* a_dst0 = (const float*)d_in[7];
    const float* b_gat0 = (const float*)d_in[8];
    const float* w_lin0 = (const float*)d_in[9];
    const float* b_lin0 = (const float*)d_in[10];
    const float* w_pre1 = (const float*)d_in[11];
    const float* b_pre1 = (const float*)d_in[12];
    const float* w_gat1 = (const float*)d_in[13];
    const float* a_src1 = (const float*)d_in[14];
    const float* a_dst1 = (const float*)d_in[15];
    const float* b_gat1 = (const float*)d_in[16];
    const float* b_gat1_ = b_gat1; (void)b_gat1_;
    const float* w_lin1 = (const float*)d_in[17];
    const float* b_lin1 = (const float*)d_in[18];
    const float* w_cls  = (const float*)d_in[19];
    const float* b_cls  = (const float*)d_in[20];

    char* ws = (char*)d_ws;
    float*    outF  = (float*)(ws + o_out);
    _Float16* in16a = (_Float16*)(ws + o_go);     /* layer-0 x_f16 aliases dead gat-out */
    _Float16* in16b = (_Float16*)(ws + o_out16);  /* layer-1 / cls f16 input */
    _Float16* wcls_t = (_Float16*)(ws + o_w + 172032);
    float* logits = (float*)d_out;

    /* layer 0: F=500 padded to 512 */
    run_layer(x, F_IN, 512, w_pre0, b_pre0, w_gat0, a_src0, a_dst0, b_gat0,
              w_lin0, b_lin0, srcI, dstI, E, ws, in16a, outF, stream);
    /* layer 1: F=64 */
    run_layer(outF, HID, HID, w_pre1, b_pre1, w_gat1, a_src1, a_dst1, b_gat1,
              w_lin1, b_lin1, srcI, dstI, E, ws, in16b, outF, stream);

    /* classifier + log_softmax */
    {
        long long tot = (long long)GN * HID;
        conv_f16_k<<<CDIV(tot, 256), 256, 0, stream>>>(outF, in16b, HID, HID, tot);
        transpose_w_k<<<CDIV(NC * HID, 256), 256, 0, stream>>>(w_cls, wcls_t, HID, NC, 0, HID, NC * HID);
        wmma_gemm_k<<<dim3(NC / 16, GN / 16), 32, 0, stream>>>(in16b, wcls_t, b_cls, logits,
                                                               HID, HID, HID, NC, 0);
        log_softmax_k<<<CDIV(GN, 256), 256, 0, stream>>>(logits, GN);
    }
}
